// RWKV7cTimeMix_56023553409279
// MI455X (gfx1250) — compile-verified, matrix operand order checked by reading, not run on hardware
//
#include <hip/hip_runtime.h>
#include <hip/hip_bf16.h>

// ---------------------------------------------------------------------------
// RWKV7 time-mix for MI455X (gfx1250, wave32, WMMA).
//   B=2, T=2048, C=1024, H=16, N=64, MISS_ROWS=128
// ---------------------------------------------------------------------------

typedef __attribute__((ext_vector_type(16))) __bf16 v16bf;
typedef __attribute__((ext_vector_type(8)))  __bf16 v8bf;
typedef __attribute__((ext_vector_type(8)))  float  v8f;
typedef __attribute__((ext_vector_type(4)))  int    v4i;

constexpr int BB = 2;
constexpr int TT = 2048;
constexpr int CC = 1024;
constexpr int HH = 16;
constexpr int NN = 64;
constexpr int MR = 128;          // MISS_ROWS
constexpr int MM = BB * TT;      // 4096 tokens

// Packed per-(token,head,lane) scan operands: one 12B load per step per lane.
struct __align__(4) ScanOp { __bf16 q, w, k, z, b, v; };

// ---- gfx1250 async global->LDS copy path (guarded; falls back cleanly) ----
#if defined(__HIP_DEVICE_COMPILE__) &&                                        \
    __has_builtin(__builtin_amdgcn_global_load_async_to_lds_b128) &&          \
    __has_builtin(__builtin_amdgcn_s_wait_asynccnt)
#define USE_ASYNC_LDS 1
#else
#define USE_ASYNC_LDS 0
#endif

#if USE_ASYNC_LDS
__device__ static inline void async_copy_b128(const void* g, void* l)
{
    // Builtin signature (from clang diagnostic): AS1 v4i* src, AS3 v4i* dst,
    // imm offset, imm cpol.
    __builtin_amdgcn_global_load_async_to_lds_b128(
        (__attribute__((address_space(1))) v4i*)(g),
        (__attribute__((address_space(3))) v4i*)(l),
        0, 0);
}
#endif

// ---------------------------------------------------------------------------
// Kernel 1: token-shift input mixing -> bf16 GEMM operands
// ---------------------------------------------------------------------------
__global__ __launch_bounds__(256) void prep_kernel(
    const float* __restrict__ x, const float* __restrict__ x0,
    const float* __restrict__ dx0,
    const float* __restrict__ x_q, const float* __restrict__ x_k,
    const float* __restrict__ x_v,
    __bf16* __restrict__ Xq, __bf16* __restrict__ Xk,
    __bf16* __restrict__ Xv, __bf16* __restrict__ Xv0,
    __bf16* __restrict__ Xd)
{
    size_t idx = (size_t)blockIdx.x * 256 + threadIdx.x;
    if (idx >= (size_t)MM * CC) return;
    int    c = (int)(idx % CC);
    size_t m = idx / CC;
    int    t = (int)(m % TT);

    float xv    = x[idx];
    float xprev = (t > 0) ? x[idx - CC] : 0.0f;
    float dxp   = xprev - xv;                 // x_{t-1} - x_t (zero pad at t=0)

    Xq[idx]  = (__bf16)(xv + dxp * x_q[c]);
    Xk[idx]  = (__bf16)(xv + dxp * x_k[c]);
    Xv[idx]  = (__bf16)(xv + dxp * x_v[c]);
    Xv0[idx] = (__bf16)(x0[idx] + dx0[idx] * x_v[c]);
    if (c < MR) Xd[m * MR + c] = (__bf16)xv;  // x[:, :, :128]
}

// ---------------------------------------------------------------------------
// Kernel 2: f32 weights -> bf16; miss [128,4096] -> missT [4096,128]
// ---------------------------------------------------------------------------
__global__ __launch_bounds__(256) void wconv_kernel(
    const float* __restrict__ Wq, const float* __restrict__ Wk,
    const float* __restrict__ Wv, const float* __restrict__ Wp,
    const float* __restrict__ miss,
    __bf16* __restrict__ Wqb, __bf16* __restrict__ Wkb,
    __bf16* __restrict__ Wvb, __bf16* __restrict__ Wpb,
    __bf16* __restrict__ missT)
{
    size_t i = (size_t)blockIdx.x * 256 + threadIdx.x;
    if (i < (size_t)CC * CC) {
        Wqb[i] = (__bf16)Wq[i];
        Wkb[i] = (__bf16)Wk[i];
        Wvb[i] = (__bf16)Wv[i];
        Wpb[i] = (__bf16)Wp[i];
    }
    if (i < (size_t)MR * 4 * CC) {
        size_t r = i / (4 * CC);
        size_t n = i % (4 * CC);
        missT[n * MR + r] = (__bf16)miss[i];
    }
}

// ---------------------------------------------------------------------------
// Kernel 3: bf16 WMMA GEMM,  Y[m,n] = sum_k X[m,k] * W[n,k]  (+ resid)
//   block tile 128x128, 256 threads = 8 wave32; wave tile 64x32 (4x2 WMMA).
//   K-step 32, double-buffered LDS, async global->LDS staging when available.
// ---------------------------------------------------------------------------
template <int KDIM>
__global__ __launch_bounds__(256) void gemm_bf16_wmma(
    const __bf16* __restrict__ X, const __bf16* __restrict__ W,
    float* __restrict__ Y, const float* __restrict__ resid, int Ndim)
{
    constexpr int KT    = 32;
    constexpr int KITER = KDIM / KT;
    constexpr int LDR   = 40;            // padded row (80B, 16B aligned)

    __shared__ __align__(16) __bf16 As[2][128][LDR];
    __shared__ __align__(16) __bf16 Bs[2][128][LDR];

    const int tid    = threadIdx.x;
    const int lane   = tid & 31;
    const int wave   = tid >> 5;
    const int mBase  = blockIdx.y * 128;
    const int nBase  = blockIdx.x * 128;
    const int wm     = (wave >> 2) * 64;     // 0 | 64
    const int wn     = (wave & 3) * 32;      // 0,32,64,96
    const int laneLo = lane & 15;
    const int kbase  = ((lane >> 4) & 1) * 8; // lanes 0-15 -> K{0..7,16..23};
                                              // lanes 16-31 -> K{8..15,24..31}

    v8f acc[4][2];
    #pragma unroll
    for (int i = 0; i < 4; ++i)
        #pragma unroll
        for (int j = 0; j < 2; ++j)
            #pragma unroll
            for (int e = 0; e < 8; ++e) acc[i][j][e] = 0.0f;

    auto loadTile = [&](int buf, int k0) {
        #pragma unroll
        for (int ch = 0; ch < 2; ++ch) {
            int e  = ch * 2048 + tid * 8;
            int r  = e >> 5;        // 0..127
            int kk = e & 31;        // 0,8,16,24
            const __bf16* gA = &X[(size_t)(mBase + r) * KDIM + k0 + kk];
            const __bf16* gB = &W[(size_t)(nBase + r) * KDIM + k0 + kk];
#if USE_ASYNC_LDS
            async_copy_b128(gA, &As[buf][r][kk]);
            async_copy_b128(gB, &Bs[buf][r][kk]);
#else
            *(v8bf*)&As[buf][r][kk] = *(const v8bf*)gA;
            *(v8bf*)&Bs[buf][r][kk] = *(const v8bf*)gB;
#endif
        }
    };

    loadTile(0, 0);
    for (int ki = 0; ki < KITER; ++ki) {
#if USE_ASYNC_LDS
        __builtin_amdgcn_s_wait_asynccnt(0);   // own async copies in LDS
#endif
        __syncthreads();                       // all waves' copies visible
        const int buf = ki & 1;
        if (ki + 1 < KITER) loadTile(buf ^ 1, (ki + 1) * KT);

        v16bf afrag[4], bfrag[2];
        #pragma unroll
        for (int i = 0; i < 4; ++i) {
            const __bf16* ap = &As[buf][wm + i * 16 + laneLo][kbase];
            v8bf lo = *(const v8bf*)ap;
            v8bf hi = *(const v8bf*)(ap + 16);
            afrag[i] = __builtin_shufflevector(lo, hi, 0, 1, 2, 3, 4, 5, 6, 7,
                                               8, 9, 10, 11, 12, 13, 14, 15);
        }
        #pragma unroll
        for (int j = 0; j < 2; ++j) {
            const __bf16* bp = &Bs[buf][wn + j * 16 + laneLo][kbase];
            v8bf lo = *(const v8bf*)bp;
            v8bf hi = *(const v8bf*)(bp + 16);
            bfrag[j] = __builtin_shufflevector(lo, hi, 0, 1, 2, 3, 4, 5, 6, 7,
                                               8, 9, 10, 11, 12, 13, 14, 15);
        }
        #pragma unroll
        for (int i = 0; i < 4; ++i)
            #pragma unroll
            for (int j = 0; j < 2; ++j)
                acc[i][j] = __builtin_amdgcn_wmma_f32_16x16x32_bf16(
                    false, afrag[i], false, bfrag[j],
                    (short)0, acc[i][j], false, false);
    }

    // C/D layout: VGPR v, lanes 0-15 -> (M=v, N=lane); lanes 16-31 -> (M=8+v)
    const int rowSel = lane >> 4;
    #pragma unroll
    for (int i = 0; i < 4; ++i)
        #pragma unroll
        for (int j = 0; j < 2; ++j) {
            const int m = mBase + wm + i * 16 + rowSel * 8;
            const int n = nBase + wn + j * 16 + laneLo;
            #pragma unroll
            for (int v = 0; v < 8; ++v) {
                size_t off = (size_t)(m + v) * Ndim + n;
                float val  = acc[i][j][v];
                if (resid) val += resid[off];
                Y[off] = val;
            }
        }
}

// ---------------------------------------------------------------------------
// Kernel 4: per-(token,head) elementwise + k-normalization; bf16 rounding of
// the six scan operands (matching the reference's rounding point); packs
// them into one 12B record per (token,head,lane) for the scan.
// ---------------------------------------------------------------------------
__global__ __launch_bounds__(64) void post_gemm_kernel(
    const float* __restrict__ qf, const float* __restrict__ kf,
    const float* __restrict__ vfr, const float* __restrict__ vfirst,
    const float* __restrict__ deltas,
    const float* __restrict__ w0, const float* __restrict__ v0,
    const float* __restrict__ a0,
    ScanOp* __restrict__ sops,
    __bf16* __restrict__ vh, __bf16* __restrict__ gb)
{
    const int    g = blockIdx.x;      // m*H + h
    const int    h = g % HH;
    const size_t m = (size_t)(g / HH);
    const int    n = threadIdx.x;
    const int    c = h * NN + n;
    const size_t idx   = m * CC + c;
    const size_t dbase = m * (size_t)(4 * CC);

    const float kv = kf[idx];

    __shared__ float red[NN];
    red[n] = kv * kv;
    __syncthreads();
    #pragma unroll
    for (int s = 32; s > 0; s >>= 1) {
        if (n < s) red[n] += red[n + s];
        __syncthreads();
    }
    const float norm = sqrtf(red[0]);

    const float wd = deltas[dbase + c];
    const float vd = deltas[dbase + CC + c];
    const float ad = deltas[dbase + 2 * CC + c];
    const float gd = deltas[dbase + 3 * CC + c];

    const float xw   = -(w0[c] + wd);
    const float sp   = (xw > 20.0f) ? xw : log1pf(__expf(xw)); // softplus
    const float logw = -__expf(-sp - 0.5f);                    // log decay < 0

    const float sv = 1.0f / (1.0f + __expf(-(v0[c] + vd)));
    float vv = vfr[idx];
    vv = vv + (vfirst[idx] - vv) * sv;

    const float a  = 1.0f / (1.0f + __expf(-(a0[c] + ad)));
    const float kk = kv / fmaxf(norm, 1e-12f);

    ScanOp so;
    so.q = (__bf16)qf[idx];
    so.w = (__bf16)logw;
    so.k = (__bf16)(kv * a);
    so.z = (__bf16)(-kk);
    so.b = (__bf16)(kk * a);
    so.v = (__bf16)vv;
    sops[idx] = so;

    vh[idx] = (__bf16)vv;
    gb[idx] = (__bf16)(1.0f + gd);
}

// ---------------------------------------------------------------------------
// Kernel 5: sequential delta-rule scan. One block per (b,h); 64 threads;
// lane v owns state row S[v][0..63] in VGPRs; operands broadcast via LDS.
// Next step's packed operand is prefetched into registers while the current
// step computes, hiding global-load latency on the serial critical path.
//   S = S*diag(e^w) + (S a) b^T + v k^T ;  o = S q
// ---------------------------------------------------------------------------
__global__ __launch_bounds__(64) void scan_kernel(
    const ScanOp* __restrict__ sops, float* __restrict__ y)
{
    const int g = blockIdx.x;          // b*H + h
    const int h = g % HH;
    const int b = g / HH;
    const int v = threadIdx.x;         // value-dim row owned by this lane

    float S[NN];
    #pragma unroll
    for (int i = 0; i < NN; ++i) S[i] = 0.0f;

    __shared__ float qs[NN], es[NN], ks[NN], as_[NN], bs[NN], vs[NN];

    size_t base = ((size_t)b * TT) * CC + (size_t)h * NN;   // t = 0
    ScanOp cur  = sops[base + v];

    for (int t = 0; t < TT; ++t) {
        qs[v]  = (float)cur.q;
        es[v]  = __expf((float)cur.w);   // exp after bf16 round, as reference
        ks[v]  = (float)cur.k;
        as_[v] = (float)cur.z;           // scan's a_t = z = -kk
        bs[v]  = (float)cur.b;
        vs[v]  = (float)cur.v;
        __syncthreads();

        const size_t nbase = base + CC;
        if (t + 1 < TT) cur = sops[nbase + v];   // prefetch next step

        float sa = 0.0f;
        #pragma unroll
        for (int i = 0; i < NN; ++i) sa += S[i] * as_[i];

        const float vt = vs[v];
        float o = 0.0f;
        #pragma unroll
        for (int i = 0; i < NN; ++i) {
            float s = S[i] * es[i] + sa * bs[i] + vt * ks[i];
            S[i] = s;
            o += s * qs[i];
        }
        y[base + v] = o * 0.125f;        // * N^-0.5
        base = nbase;
        __syncthreads();                 // protect LDS for next step
    }
}

// ---------------------------------------------------------------------------
// Kernel 6: norm-ratio correction + gate -> bf16 input for output projection
// ---------------------------------------------------------------------------
__global__ __launch_bounds__(64) void post_scan_kernel(
    const float* __restrict__ y, const float* __restrict__ x,
    const __bf16* __restrict__ vh, const __bf16* __restrict__ gb,
    const float* __restrict__ r_k, __bf16* __restrict__ Yg)
{
    const int    g = blockIdx.x;
    const int    h = g % HH;
    const size_t m = (size_t)(g / HH);
    const int    n = threadIdx.x;
    const int    c = h * NN + n;
    const size_t idx = m * CC + c;

    const float xv = x[idx];
    const float vv = (float)vh[idx];

    __shared__ float rx[NN], rv[NN];
    rx[n] = xv * xv;
    rv[n] = vv * vv;
    __syncthreads();
    #pragma unroll
    for (int s = 32; s > 0; s >>= 1) {
        if (n < s) { rx[n] += rx[n + s]; rv[n] += rv[n + s]; }
        __syncthreads();
    }
    const float xnorm = sqrtf(rx[0]);
    const float vnorm = sqrtf(rv[0]);

    const float sig = 1.0f / (1.0f + __expf(-r_k[h * NN + n]));
    const float u   = (1.0f - xnorm / (vnorm + 1e-12f)) * sig;

    const float yv = y[idx] + vv * u;
    Yg[idx] = (__bf16)(yv * (float)gb[idx]);
}

// ---------------------------------------------------------------------------
// Host launcher
// ---------------------------------------------------------------------------
extern "C" void kernel_launch(void* const* d_in, const int* in_sizes, int n_in,
                              void* d_out, int out_size, void* d_ws,
                              size_t ws_size, hipStream_t stream)
{
    (void)in_sizes; (void)n_in; (void)out_size; (void)ws_size;

    const float* residual = (const float*)d_in[0];
    const float* x        = (const float*)d_in[1];
    const float* x0       = (const float*)d_in[2];
    const float* dx0      = (const float*)d_in[3];
    const float* Wq       = (const float*)d_in[4];
    const float* Wk       = (const float*)d_in[5];
    const float* Wv       = (const float*)d_in[6];
    const float* Wproj    = (const float*)d_in[7];
    const float* x_q      = (const float*)d_in[8];
    const float* x_k      = (const float*)d_in[9];
    const float* x_v      = (const float*)d_in[10];
    const float* v0       = (const float*)d_in[11];
    const float* w0       = (const float*)d_in[12];
    const float* a0       = (const float*)d_in[13];
    const float* miss     = (const float*)d_in[14];
    const float* r_k      = (const float*)d_in[15];
    float*       out      = (float*)d_out;

    // ---- workspace layout (256B aligned bump allocator) ----
    char*  ws  = (char*)d_ws;
    size_t off = 0;
    auto alloc = [&](size_t bytes) -> void* {
        void* p = ws + off;
        off += (bytes + 255) & ~(size_t)255;
        return p;
    };
    const size_t MC_BF = (size_t)MM * CC * sizeof(__bf16);    // 8 MB
    const size_t MC_F  = (size_t)MM * CC * sizeof(float);     // 16 MB
    const size_t W_BF  = (size_t)CC * CC * sizeof(__bf16);    // 2 MB
    const size_t XD_BF = (size_t)MM * MR * sizeof(__bf16);    // 1 MB
    const size_t DL_F  = (size_t)MM * 4 * CC * sizeof(float); // 64 MB
    const size_t SO_SZ = (size_t)MM * CC * sizeof(ScanOp);    // 48 MB

    __bf16* Xq    = (__bf16*)alloc(MC_BF);
    __bf16* Xk    = (__bf16*)alloc(MC_BF);
    __bf16* Xv    = (__bf16*)alloc(MC_BF);
    __bf16* Xv0   = (__bf16*)alloc(MC_BF);
    __bf16* Xd    = (__bf16*)alloc(XD_BF);
    __bf16* Wqb   = (__bf16*)alloc(W_BF);
    __bf16* Wkb   = (__bf16*)alloc(W_BF);
    __bf16* Wvb   = (__bf16*)alloc(W_BF);
    __bf16* Wpb   = (__bf16*)alloc(W_BF);
    __bf16* missT = (__bf16*)alloc(XD_BF);
    float*  qf    = (float*)alloc(MC_F);
    float*  kf    = (float*)alloc(MC_F);
    float*  vfr   = (float*)alloc(MC_F);
    float*  vfirst= (float*)alloc(MC_F);
    float*  deltas= (float*)alloc(DL_F);
    ScanOp* sops  = (ScanOp*)alloc(SO_SZ);
    __bf16* gb    = (__bf16*)alloc(MC_BF);
    // aliases over buffers whose last reader precedes the new writer:
    __bf16* vh  = Xv0;       // Xv0 consumed by v_first GEMM before post_gemm
    float*  y   = qf;        // qf consumed by post_gemm before scan writes
    __bf16* Yg  = (__bf16*)kf;

    // ---- 1) input prep + weight conversion ----
    {
        const size_t total = (size_t)MM * CC;
        prep_kernel<<<dim3((unsigned)((total + 255) / 256)), dim3(256), 0, stream>>>(
            x, x0, dx0, x_q, x_k, x_v, Xq, Xk, Xv, Xv0, Xd);
        const size_t wtot = (size_t)CC * CC;
        wconv_kernel<<<dim3((unsigned)((wtot + 255) / 256)), dim3(256), 0, stream>>>(
            Wq, Wk, Wv, Wproj, miss, Wqb, Wkb, Wvb, Wpb, missT);
    }

    // ---- 2) WMMA GEMMs: q, k, v, v_first (K=1024, N=1024), deltas (K=128, N=4096)
    {
        dim3 grid(CC / 128, MM / 128), blk(256);
        gemm_bf16_wmma<CC><<<grid, blk, 0, stream>>>(Xq,  Wqb, qf,     (const float*)nullptr, CC);
        gemm_bf16_wmma<CC><<<grid, blk, 0, stream>>>(Xk,  Wkb, kf,     (const float*)nullptr, CC);
        gemm_bf16_wmma<CC><<<grid, blk, 0, stream>>>(Xv,  Wvb, vfr,    (const float*)nullptr, CC);
        gemm_bf16_wmma<CC><<<grid, blk, 0, stream>>>(Xv0, Wvb, vfirst, (const float*)nullptr, CC);
        dim3 gridD(4 * CC / 128, MM / 128);
        gemm_bf16_wmma<MR><<<gridD, blk, 0, stream>>>(Xd, missT, deltas, (const float*)nullptr, 4 * CC);
    }

    // ---- 3) elementwise + k-norm -> packed bf16 scan operands ----
    post_gemm_kernel<<<dim3(MM * HH), dim3(64), 0, stream>>>(
        qf, kf, vfr, vfirst, deltas, w0, v0, a0, sops, vh, gb);

    // ---- 4) sequential scan (state in VGPRs) ----
    scan_kernel<<<dim3(BB * HH), dim3(64), 0, stream>>>(sops, y);

    // ---- 5) norm correction + gate ----
    post_scan_kernel<<<dim3(MM * HH), dim3(64), 0, stream>>>(y, x, vh, gb, r_k, Yg);

    // ---- 6) output projection + residual -> d_out ----
    {
        dim3 grid(CC / 128, MM / 128), blk(256);
        gemm_bf16_wmma<CC><<<grid, blk, 0, stream>>>(Yg, Wpb, out, residual, CC);
    }
}
